// Net_30537217474606
// MI455X (gfx1250) — matrix operand order repaired and checked by reading, not compiled, not run on hardware
//
#include <hip/hip_runtime.h>

// ---------------------------------------------------------------------------
// MI455X / gfx1250 implementation.
// Key algebraic optimization: pcd[b,c,i,j] = (c<1280 ? yc_s[b,c,i] : yc_t[b,c-1280,j])
// so the pn/pe intro 1x1-convs decompose into small GEMMs on yc_s/yc_t (+pcc)
// followed by a broadcast-add.  We never materialize the (B,2560,48,48) pcd or
// the (B,4608,48,48) pe input (saves ~264MB of traffic and ~270 GFLOP).
// Heavy GEMMs run on v_wmma_f32_16x16x32_bf16 with double-buffered LDS bf16
// tiles laid out so every fragment is two ds_load_b128.
// ---------------------------------------------------------------------------

typedef __attribute__((ext_vector_type(16))) __bf16 v16bf;
typedef __attribute__((ext_vector_type(8)))  __bf16 v8bf;
typedef __attribute__((ext_vector_type(8)))  float  v8f;

// ---------------------------------------------------------------------------
// GEMM:  Y[b] = W @ X[b] (+ bias) (+ Res[b])
//   W    : (Cout, ldw) row-major (ldw lets us slice columns of a wider W)
//   X[b] : (Cin, M) row-major at X + b*sXb
//   Y[b] : (Cout, M) row-major at Y + b*sYb ; Res same layout (nullable)
// Block = 256 threads = 8 wave32; block tile 128(out) x 64(M);
// per-wave 16x64 (A fragment reused across 4 wmma).
// LDS layouts:
//   Wt[r][pos(k)]  with pos(k) = k (natural k order)
//       -> lane-group g reads its 16 A elements contiguously at pos g*16.
//   Xt[m][k]       (transposed) -> B fragment reads 16 contiguous k at g*16.
// Rows padded to 40 bf16 (80B, 16B-multiple) so v8bf loads are aligned and
// 16-lane groups hit 16 distinct bank quads.
// Uniform full-tile fast paths keep the hot staging loops free of per-lane
// predication (no exec churn); guarded path only runs on edge tiles.
// ---------------------------------------------------------------------------
__global__ void __launch_bounds__(256)
gemm_wmma_kernel(const float* __restrict__ W, const float* __restrict__ bias,
                 const float* __restrict__ X, const float* __restrict__ Res,
                 float* __restrict__ Y,
                 int Cout, int Cin, int M, int ldw, long sXb, long sYb)
{
  __shared__ alignas(16) __bf16 Wt[2][128][40];
  __shared__ alignas(16) __bf16 Xt[2][64][40];

  const int tid  = threadIdx.x;
  const int wv   = tid >> 5;          // wave 0..7
  const int lane = tid & 31;
  const int l15  = lane & 15;
  const int g    = lane >> 4;         // lane group 0/1
  const int b    = blockIdx.z;
  const int obase0 = blockIdx.y * 128;
  const int mbase0 = blockIdx.x * 64;

  const float* Xb = X + (long)b * sXb;
  const bool rowsFull = (obase0 + 128 <= Cout);
  const bool colsFull = (mbase0 + 64 <= M);

  auto stageW = [&](int buf, int k) {
    if (rowsFull && (k + 32 <= Cin)) {
      // uniform fast path: no per-lane guards
      #pragma unroll
      for (int idx = tid; idx < 128 * 32; idx += 256) {
        int r = idx >> 5, c = idx & 31;
        float v = W[(long)(obase0 + r) * ldw + (k + c)];
        int pos = c;
        Wt[buf][r][pos] = (__bf16)v;
      }
    } else {
      #pragma unroll
      for (int idx = tid; idx < 128 * 32; idx += 256) {
        int r = idx >> 5, c = idx & 31;
        int gr = obase0 + r, gc = k + c;
        float v = (gr < Cout && gc < Cin) ? W[(long)gr * ldw + gc] : 0.0f;
        int pos = c;
        Wt[buf][r][pos] = (__bf16)v;
      }
    }
  };
  auto stageX = [&](int buf, int k) {
    if (colsFull && (k + 32 <= Cin)) {
      // uniform fast path
      #pragma unroll
      for (int idx = tid; idx < 64 * 32; idx += 256) {
        int r = idx >> 6, c = idx & 63;         // r = local k, c = local m
        float v = Xb[(long)(k + r) * M + (mbase0 + c)];
        Xt[buf][c][r] = (__bf16)v;              // transposed store
      }
    } else {
      #pragma unroll
      for (int idx = tid; idx < 64 * 32; idx += 256) {
        int r = idx >> 6, c = idx & 63;
        int gk = k + r, gm = mbase0 + c;
        float v = (gk < Cin && gm < M) ? Xb[(long)gk * M + gm] : 0.0f;
        Xt[buf][c][r] = (__bf16)v;
      }
    }
  };

  v8f acc[4] = {};
  const int nk = (Cin + 31) >> 5;

  stageW(0, 0);
  stageX(0, 0);

  for (int kt = 0; kt < nk; ++kt) {
    __syncthreads();                     // tile kt staged; prev reads finished
    const int buf = kt & 1;
    if (kt + 1 < nk) {
      int k2 = (kt + 1) * 32;
      if (kt + 2 < nk)                   // speculative L2 prefetch 2 tiles ahead
        __builtin_prefetch(Xb + (long)(kt + 2) * 32 * M + mbase0, 0, 1);
      stageW(buf ^ 1, k2);
      stageX(buf ^ 1, k2);
    }

    // ---- A fragment: two aligned v8bf LDS loads --------------------------
    const __bf16* wrow = &Wt[buf][wv * 16 + l15][g * 16];
    v8bf alo = *(const v8bf*)(wrow);
    v8bf ahi = *(const v8bf*)(wrow + 8);
    v16bf a = __builtin_shufflevector(alo, ahi, 0, 1, 2, 3, 4, 5, 6, 7,
                                      8, 9, 10, 11, 12, 13, 14, 15);

    // ---- load all 4 B fragments first (distinct regs -> staggered waits) -
    v16bf bf[4];
    #pragma unroll
    for (int t = 0; t < 4; ++t) {
      const __bf16* xrow = &Xt[buf][t * 16 + l15][g * 16];
      v8bf blo = *(const v8bf*)(xrow);
      v8bf bhi = *(const v8bf*)(xrow + 8);
      bf[t] = __builtin_shufflevector(blo, bhi, 0, 1, 2, 3, 4, 5, 6, 7,
                                      8, 9, 10, 11, 12, 13, 14, 15);
    }
    // ---- 4 wmma, A reused -----------------------------------------------
    #pragma unroll
    for (int t = 0; t < 4; ++t)
      acc[t] = __builtin_amdgcn_wmma_f32_16x16x32_bf16(
          false, a, false, bf[t], (short)0, acc[t], false, false);
  }

  // ---- epilogue: bias / residual / store --------------------------------
  const float* Rb = Res ? Res + (long)b * sYb : nullptr;
  float* Yb = Y + (long)b * sYb;
  if (rowsFull && colsFull) {
    #pragma unroll
    for (int t = 0; t < 4; ++t) {
      int m = mbase0 + t * 16 + l15;
      #pragma unroll
      for (int r = 0; r < 8; ++r) {
        int o = obase0 + wv * 16 + r + 8 * g;   // C/D ISA layout
        float v = acc[t][r];
        if (bias) v += bias[o];
        if (Rb)   v += Rb[(long)o * M + m];
        Yb[(long)o * M + m] = v;
      }
    }
  } else {
    #pragma unroll
    for (int t = 0; t < 4; ++t) {
      int m = mbase0 + t * 16 + l15;
      if (m < M) {
        #pragma unroll
        for (int r = 0; r < 8; ++r) {
          int o = obase0 + wv * 16 + r + 8 * g;
          if (o < Cout) {
            float v = acc[t][r];
            if (bias) v += bias[o];
            if (Rb)   v += Rb[(long)o * M + m];
            Yb[(long)o * M + m] = v;
          }
        }
      }
    }
  }
}

// ---------------------------------------------------------------------------
// Bilinear feature sampling (jax grid_sample equivalent), U layout (B,1536,N)
// ---------------------------------------------------------------------------
__global__ void align_kernel(const float* __restrict__ f0, const float* __restrict__ f1,
                             const float* __restrict__ f2, const float* __restrict__ f3,
                             const float* __restrict__ f4, const float* __restrict__ f5,
                             const float* __restrict__ P, float* __restrict__ U, int Np)
{
  const int n = blockIdx.x;
  const int b = blockIdx.y;
  const float px = P[((long)b * Np + n) * 2 + 0];
  const float py = P[((long)b * Np + n) * 2 + 1];
  const float* fp[6] = {f0, f1, f2, f3, f4, f5};
  const int Cs[6]  = {64, 64, 128, 256, 512, 512};
  const int Hs[6]  = {128, 64, 32, 16, 8, 1};
  const int Off[6] = {0, 64, 128, 256, 512, 1024};

  for (int c = threadIdx.x; c < 1536; c += blockDim.x) {
    int l = 0;
    #pragma unroll
    for (int q = 0; q < 5; ++q) if (c >= Off[q + 1]) l = q + 1;
    int cl = c - Off[l];
    int H = Hs[l], Wd = Hs[l];
    float ix = (px * (2.0f / 256.0f) * (float)Wd - 1.0f) * 0.5f;
    float iy = (py * (2.0f / 256.0f) * (float)H  - 1.0f) * 0.5f;
    ix = fminf(fmaxf(ix, 0.0f), (float)(Wd - 1));
    iy = fminf(fmaxf(iy, 0.0f), (float)(H - 1));
    float x0f = floorf(ix), y0f = floorf(iy);
    int x0 = (int)x0f, y0 = (int)y0f;
    int x1 = min(x0 + 1, Wd - 1), y1 = min(y0 + 1, H - 1);
    float wx = ix - x0f, wy = iy - y0f;
    const float* f = fp[l] + (long)(b * Cs[l] + cl) * H * Wd;
    float v00 = f[y0 * Wd + x0], v01 = f[y0 * Wd + x1];
    float v10 = f[y1 * Wd + x0], v11 = f[y1 * Wd + x1];
    float v = v00 * (1.0f - wx) * (1.0f - wy) + v01 * wx * (1.0f - wy)
            + v10 * (1.0f - wx) * wy          + v11 * wx * wy;
    U[((long)b * 1536 + c) * Np + n] = v;
  }
}

// ---------------------------------------------------------------------------
// d = relu(gamma*(x-mean)*rsqrt(var+eps)+beta), x layout (B,C,M)
// ---------------------------------------------------------------------------
__global__ void bnrelu_kernel(const float* __restrict__ x,
                              const float* __restrict__ ga, const float* __restrict__ be,
                              const float* __restrict__ mu, const float* __restrict__ va,
                              float* __restrict__ d, int C, long M, long total)
{
  for (long i = (long)blockIdx.x * blockDim.x + threadIdx.x; i < total;
       i += (long)gridDim.x * blockDim.x) {
    int c = (int)((i / M) % C);
    float v = (x[i] - mu[c]) * rsqrtf(va[c] + 1e-5f);
    v = v * ga[c] + be[c];
    d[i] = fmaxf(v, 0.0f);
  }
}

// ---------------------------------------------------------------------------
// Rank-structured intro epilogue:
// Y[b,c,i*Nn+j] = (Ap?Ap[b,c]:0) + As[b,c,i] + At[b,c,j] + bias[c]
// ---------------------------------------------------------------------------
__global__ void bcastadd_kernel(const float* __restrict__ Ap, const float* __restrict__ As,
                                const float* __restrict__ At, const float* __restrict__ bias,
                                float* __restrict__ Y, int C, int Nn)
{
  const int b = blockIdx.z;
  const long total = (long)C * Nn * Nn;
  for (long i = (long)blockIdx.x * blockDim.x + threadIdx.x; i < total;
       i += (long)gridDim.x * blockDim.x) {
    int j = (int)(i % Nn);
    long t = i / Nn;
    int ii = (int)(t % Nn);
    int c  = (int)(t / Nn);
    float v = As[((long)b * C + c) * Nn + ii] + At[((long)b * C + c) * Nn + j] + bias[c];
    if (Ap) v += Ap[(long)b * C + c];
    Y[(long)b * total + i] = v;
  }
}

// ---------------------------------------------------------------------------
// pcc[b,c] = max over (i,j) of X[b,c,i,j] * ((i<ns_row[b]) & (j<ns_col[b]))
// ---------------------------------------------------------------------------
__global__ void maskedmax_kernel(const float* __restrict__ X,
                                 const int* __restrict__ ns_row, const int* __restrict__ ns_col,
                                 float* __restrict__ pcc, int C, int Nn)
{
  __shared__ float red[256];
  const int c = blockIdx.x, b = blockIdx.y;
  const int nr = ns_row[b], nc = ns_col[b];
  const float* xb = X + ((long)b * C + c) * Nn * Nn;
  float mx = -__builtin_inff();
  for (int i = threadIdx.x; i < Nn * Nn; i += blockDim.x) {
    int ii = i / Nn, j = i % Nn;
    float v = xb[i];
    v = (ii < nr && j < nc) ? v : 0.0f;   // val * mask
    mx = fmaxf(mx, v);
  }
  red[threadIdx.x] = mx;
  __syncthreads();
  for (int s = 128; s > 0; s >>= 1) {
    if ((int)threadIdx.x < s) red[threadIdx.x] = fmaxf(red[threadIdx.x], red[threadIdx.x + s]);
    __syncthreads();
  }
  if (threadIdx.x == 0) pcc[(long)b * C + c] = red[0];
}

// ---------------------------------------------------------------------------
// pf input: (B,3,N) = [Px/256, Py/256, const]
// ---------------------------------------------------------------------------
__global__ void pfin_kernel(const float* __restrict__ P, float third,
                            float* __restrict__ out, int Np)
{
  int b = blockIdx.x;
  for (int n = threadIdx.x; n < Np; n += blockDim.x) {
    out[((long)b * 3 + 0) * Np + n] = P[((long)b * Np + n) * 2 + 0] * (1.0f / 256.0f);
    out[((long)b * 3 + 1) * Np + n] = P[((long)b * Np + n) * 2 + 1] * (1.0f / 256.0f);
    out[((long)b * 3 + 2) * Np + n] = third;
  }
}

// ---------------------------------------------------------------------------
// Sinkhorn: 20 alternating log-normalizations on (48x48) in LDS, one block/batch
// ---------------------------------------------------------------------------
__global__ void sinkhorn_kernel(const float* __restrict__ sim1, const float* __restrict__ sim2,
                                const int* __restrict__ ns_src, const int* __restrict__ ns_tgt,
                                float* __restrict__ out, int Nn)
{
  __shared__ float ls[48][49];
  const int b = blockIdx.x;
  const int nsrc = ns_src[b], ntgt = ns_tgt[b];
  const float NEG = -1e30f;

  for (int i = threadIdx.x; i < Nn * Nn; i += blockDim.x) {
    int r = i / Nn, c = i % Nn;
    float s = sim1[(long)b * Nn * Nn + i] + sim2[(long)b * Nn * Nn + i];
    bool dummy = (r >= nsrc) && (r < ntgt);
    if (dummy && (c < ntgt)) s = -100.0f;
    bool mask = (r < ntgt) && (c < ntgt);
    ls[r][c] = mask ? s * 20.0f : NEG;   // s / TAU, TAU = 0.05
  }
  __syncthreads();

  for (int it = 0; it < 20; ++it) {
    int t = threadIdx.x;
    if (t < Nn) {
      if ((it & 1) == 0) {               // axis=2: per-row lse over j
        float m = NEG;
        for (int j = 0; j < Nn; ++j) m = fmaxf(m, ls[t][j]);
        float ssum = 0.0f;
        for (int j = 0; j < Nn; ++j) ssum += __expf(ls[t][j] - m);
        float lse = m + __logf(ssum);
        bool rok = t < ntgt;
        for (int j = 0; j < Nn; ++j)
          ls[t][j] = (rok && (j < ntgt)) ? (ls[t][j] - lse) : NEG;
      } else {                           // axis=1: per-column lse over i
        float m = NEG;
        for (int i = 0; i < Nn; ++i) m = fmaxf(m, ls[i][t]);
        float ssum = 0.0f;
        for (int i = 0; i < Nn; ++i) ssum += __expf(ls[i][t] - m);
        float lse = m + __logf(ssum);
        bool cok = t < ntgt;
        for (int i = 0; i < Nn; ++i)
          ls[i][t] = ((i < ntgt) && cok) ? (ls[i][t] - lse) : NEG;
      }
    }
    __syncthreads();
  }

  for (int i = threadIdx.x; i < Nn * Nn; i += blockDim.x) {
    int r = i / Nn, c = i % Nn;
    bool mask = (r < ntgt) && (c < ntgt);
    out[(long)b * Nn * Nn + i] = mask ? __expf(ls[r][c]) : 0.0f;
  }
}

// ---------------------------------------------------------------------------
// Host orchestration
// ---------------------------------------------------------------------------
extern "C" void kernel_launch(void* const* d_in, const int* in_sizes, int n_in,
                              void* d_out, int out_size, void* d_ws, size_t ws_size,
                              hipStream_t stream)
{
  (void)in_sizes; (void)out_size; (void)ws_size;
  if (n_in < 56) return;

  const int B = 4, N = 48, NN = N * N;

  const float *fs[6], *ft[6];
  for (int i = 0; i < 6; ++i) { fs[i] = (const float*)d_in[2 * i]; ft[i] = (const float*)d_in[2 * i + 1]; }
  const float* P_src = (const float*)d_in[12];
  const float* P_tgt = (const float*)d_in[13];
  const int*  ns_src = (const int*)d_in[14];
  const int*  ns_tgt = (const int*)d_in[15];

  struct ResP { const float *iw,*ib,*cw,*cb,*g,*be,*mu,*va,*ow,*ob; };
  auto getp = [&](int k) {
    ResP r;
    r.iw=(const float*)d_in[k+0]; r.ib=(const float*)d_in[k+1];
    r.cw=(const float*)d_in[k+2]; r.cb=(const float*)d_in[k+3];
    r.g =(const float*)d_in[k+4]; r.be=(const float*)d_in[k+5];
    r.mu=(const float*)d_in[k+6]; r.va=(const float*)d_in[k+7];
    r.ow=(const float*)d_in[k+8]; r.ob=(const float*)d_in[k+9];
    return r;
  };
  ResP cls = getp(16), pf = getp(26), pn = getp(36), pe = getp(46);

  // --- workspace carve ----------------------------------------------------
  float* w = (float*)d_ws;
  auto carve = [&](long nf) { float* p = w; w += nf; return p; };
  float* Us   = carve((long)B * 1536 * N);
  float* Ut   = carve((long)B * 1536 * N);
  float* Xa   = carve((long)B * 2048 * N);
  float* Da   = carve((long)B * 2048 * N);
  float* ycs  = carve((long)B * 1280 * N);
  float* yct  = carve((long)B * 1280 * N);
  float* pfin = carve((long)B * 3 * N);
  float* As   = carve((long)B * 2048 * N);
  float* At   = carve((long)B * 2048 * N);
  float* Ap   = carve((long)B * 2048);
  float* pcc  = carve((long)B * 2048);
  float* X1   = carve((long)B * 2048 * NN);   // 75.5 MB
  float* Dbig = carve((long)B * 2048 * NN);   // 75.5 MB
  float* sim1 = carve((long)B * NN);
  float* sim2 = carve((long)B * NN);

  auto gemm = [&](const float* W, int ldw, const float* bias,
                  const float* X, long sXb, const float* Res,
                  float* Y, long sYb, int Cout, int Cin, int M) {
    dim3 grid((M + 63) / 64, (Cout + 127) / 128, B);
    gemm_wmma_kernel<<<grid, 256, 0, stream>>>(W, bias, X, Res, Y, Cout, Cin, M, ldw, sXb, sYb);
  };

  // --- feature sampling ---------------------------------------------------
  align_kernel<<<dim3(N, B), 256, 0, stream>>>(fs[0],fs[1],fs[2],fs[3],fs[4],fs[5], P_src, Us, N);
  align_kernel<<<dim3(N, B), 256, 0, stream>>>(ft[0],ft[1],ft[2],ft[3],ft[4],ft[5], P_tgt, Ut, N);

  // --- cls rescls: 1536 -> 2048 -> (2 chorus) -> 1024, into yc channels 256..1279
  auto run_cls = [&](const float* U, float* yc) {
    gemm(cls.iw, 1536, cls.ib, U, (long)1536 * N, nullptr, Xa, (long)2048 * N, 2048, 1536, N);
    for (int i = 0; i < 2; ++i) {
      bnrelu_kernel<<<dim3(256), 256, 0, stream>>>(Xa, cls.g + i*2048, cls.be + i*2048,
          cls.mu + i*2048, cls.va + i*2048, Da, 2048, (long)N, (long)B * 2048 * N);
      gemm(cls.cw + (long)i*2048*2048, 2048, cls.cb + i*2048, Da, (long)2048 * N,
           Xa, Xa, (long)2048 * N, 2048, 2048, N);
    }
    gemm(cls.ow, 2048, cls.ob, Xa, (long)2048 * N, nullptr, yc + 256 * N, (long)1280 * N, 1024, 2048, N);
  };
  run_cls(Us, ycs);
  run_cls(Ut, yct);

  // --- pf rescls: 3 -> 128 -> 256, into yc channels 0..255 ---------------
  auto run_pf = [&](const float* P, float third, float* yc) {
    pfin_kernel<<<dim3(B), 64, 0, stream>>>(P, third, pfin, N);
    gemm(pf.iw, 3, pf.ib, pfin, (long)3 * N, nullptr, Xa, (long)128 * N, 128, 3, N);
    bnrelu_kernel<<<dim3(64), 256, 0, stream>>>(Xa, pf.g, pf.be, pf.mu, pf.va,
        Da, 128, (long)N, (long)B * 128 * N);
    gemm(pf.cw, 128, pf.cb, Da, (long)128 * N, Xa, Xa, (long)128 * N, 128, 128, N);
    gemm(pf.ow, 128, pf.ob, Xa, (long)128 * N, nullptr, yc, (long)1280 * N, 256, 128, N);
  };
  run_pf(P_src, 0.0f, ycs);
  run_pf(P_tgt, 1.0f, yct);

  // --- _points (rank-decomposed intros) -----------------------------------
  auto run_points = [&](const float* ycS, const float* ycT,
                        const int* nsS, const int* nsT, float* simOut) {
    // pn intro: X1[c,i,j] = W[:, :1280]@ycS |_i + W[:,1280:]@ycT |_j + b
    gemm(pn.iw,        2560, nullptr, ycS, (long)1280 * N, nullptr, As, (long)2048 * N, 2048, 1280, N);
    gemm(pn.iw + 1280, 2560, nullptr, ycT, (long)1280 * N, nullptr, At, (long)2048 * N, 2048, 1280, N);
    bcastadd_kernel<<<dim3(1024, 1, B), 256, 0, stream>>>(nullptr, As, At, pn.ib, X1, 2048, N);
    // pn chorus + outro
    bnrelu_kernel<<<dim3(2048), 256, 0, stream>>>(X1, pn.g, pn.be, pn.mu, pn.va,
        Dbig, 2048, (long)NN, (long)B * 2048 * NN);
    gemm(pn.cw, 2048, pn.cb, Dbig, (long)2048 * NN, X1, X1, (long)2048 * NN, 2048, 2048, NN);
    gemm(pn.ow, 2048, pn.ob, X1, (long)2048 * NN, nullptr, Dbig, (long)2048 * NN, 2048, 2048, NN);
    maskedmax_kernel<<<dim3(2048, B), 256, 0, stream>>>(Dbig, nsS, nsT, pcc, 2048, N);
    // pe intro: cols 0..2047 x pcc (bcast), 2048..3327 x ycS, 3328..4607 x ycT
    gemm(pe.iw,        4608, nullptr, pcc, (long)2048,     nullptr, Ap, (long)2048,     2048, 2048, 1);
    gemm(pe.iw + 2048, 4608, nullptr, ycS, (long)1280 * N, nullptr, As, (long)2048 * N, 2048, 1280, N);
    gemm(pe.iw + 3328, 4608, nullptr, ycT, (long)1280 * N, nullptr, At, (long)2048 * N, 2048, 1280, N);
    bcastadd_kernel<<<dim3(1024, 1, B), 256, 0, stream>>>(Ap, As, At, pe.ib, X1, 2048, N);
    for (int i = 0; i < 2; ++i) {
      bnrelu_kernel<<<dim3(2048), 256, 0, stream>>>(X1, pe.g + i*2048, pe.be + i*2048,
          pe.mu + i*2048, pe.va + i*2048, Dbig, 2048, (long)NN, (long)B * 2048 * NN);
      gemm(pe.cw + (long)i*2048*2048, 2048, pe.cb + i*2048, Dbig, (long)2048 * NN,
           X1, X1, (long)2048 * NN, 2048, 2048, NN);
    }
    gemm(pe.ow, 2048, pe.ob, X1, (long)2048 * NN, nullptr, simOut, (long)NN, 1, 2048, NN);
  };
  run_points(ycs, yct, ns_src, ns_tgt, sim1);
  run_points(yct, ycs, ns_tgt, ns_src, sim2);

  // --- Sinkhorn -> d_out (B,48,48) ----------------------------------------
  sinkhorn_kernel<<<dim3(B), 64, 0, stream>>>(sim1, sim2, ns_src, ns_tgt, (float*)d_out, N);
}
